// BahdanauAttention_45165876085058
// MI455X (gfx1250) — compile-verified, matrix operand order checked by reading, not run on hardware
//
#include <hip/hip_runtime.h>

typedef __attribute__((ext_vector_type(16))) _Float16     v16h;
typedef __attribute__((ext_vector_type(8)))  float        v8f;
typedef __attribute__((ext_vector_type(4)))  float        f32x4;
typedef __attribute__((ext_vector_type(4)))  unsigned int u32x4;
typedef __attribute__((ext_vector_type(8)))  int          i32x8;
typedef __attribute__((ext_vector_type(4)))  int          i32x4;

#define B_   4
#define TE_  512
#define TD_  256
#define DE_  512
#define U_   128

// ---------------------------------------------------------------------------
// WMMA f16 GEMM:  P[M,N] = X[M,K] @ W[K,N] (+ bias[N]),  K compile-time.
// Block = 256 threads (8 wave32), block tile = 128(M) x 16(N).
// The 32k x 16n B panel is gathered once per k-step by the whole block,
// converted f32->f16, and stored in LDS in WMMA *fragment order*
// ([lane][16 halfs]) so each wave reads its B fragment as one v16h ds load.
// A fragments are contiguous float4 global loads per lane:
//   a[0..7]  = X[m, kb + 8h + 0..7],  a[8..15] = X[m, kb + 16 + 8h + 0..7]
// C/D f32: VGPR r -> (m = r + 8h, n = lane&15).
// ---------------------------------------------------------------------------
template <int K>
__global__ void gemm_wmma_f16(const float* __restrict__ X,
                              const float* __restrict__ W,
                              const float* __restrict__ bias,
                              float* __restrict__ P,
                              int N,
                              size_t strideXb, size_t strideWb, size_t stridePb) {
  X += (size_t)blockIdx.z * strideXb;
  W += (size_t)blockIdx.z * strideWb;
  P += (size_t)blockIdx.z * stridePb;

  __shared__ __align__(32) _Float16 Bfrag[32 * 16];  // fragment-ordered B panel

  const int tid   = threadIdx.x;
  const int wid   = tid >> 5;
  const int lane  = tid & 31;
  const int c16   = lane & 15;
  const int hsel  = lane >> 4;
  const int mBase = blockIdx.x * 128 + wid * 16;
  const int nBase = blockIdx.y * 16;
  const int row   = mBase + c16;   // A row held by this lane
  const int ncol  = nBase + c16;   // output column held by this lane

  v8f acc = {};

#pragma unroll 4
  for (int kb = 0; kb < K; kb += 32) {
    __syncthreads();   // previous iteration's Bfrag reads complete
    // ---- cooperative B stage: 512 fragment slots, 2 per thread ----
    {
      int pos = tid * 2;
#pragma unroll
      for (int j = 0; j < 2; ++j, ++pos) {
        const int l    = pos >> 4;          // fragment lane
        const int e    = pos & 15;          // fragment element
        const int koff = (e & 7) + 8 * (l >> 4) + ((e >> 3) << 4);
        const int n    = l & 15;
        Bfrag[pos] = (_Float16)W[(size_t)(kb + koff) * N + (nBase + n)];
      }
    }
    __syncthreads();

    // ---- A fragment: two contiguous 8-float runs of this lane's row ----
    const float* ap = X + (size_t)row * K + kb + 8 * hsel;
    f32x4 a0 = *(const f32x4*)(ap + 0);
    f32x4 a1 = *(const f32x4*)(ap + 4);
    f32x4 a2 = *(const f32x4*)(ap + 16);
    f32x4 a3 = *(const f32x4*)(ap + 20);
    v16h a;
#pragma unroll
    for (int i = 0; i < 4; ++i) {
      a[i]      = (_Float16)a0[i];
      a[4 + i]  = (_Float16)a1[i];
      a[8 + i]  = (_Float16)a2[i];
      a[12 + i] = (_Float16)a3[i];
    }

    // ---- B fragment: one contiguous 32-byte LDS read ----
    v16h b = *(const v16h*)&Bfrag[lane * 16];

    acc = __builtin_amdgcn_wmma_f32_16x16x32_f16(
        false, a, false, b, (short)0, acc, false, false);
  }

  const float bn = bias ? bias[ncol] : 0.0f;
#pragma unroll
  for (int r = 0; r < 8; ++r) {
    const int m = mBase + r + 8 * hsel;
    P[(size_t)m * N + ncol] = acc[r] + bn;
  }
}

// ---------------------------------------------------------------------------
// Stage 3+4: score -> softmax. One 256-thread block per (b,t).
// dec_p row is brought into LDS by the Tensor Data Mover (128x1 f32 tile).
// ---------------------------------------------------------------------------
__device__ inline float waveMax(float v) {
#pragma unroll
  for (int off = 16; off > 0; off >>= 1) v = fmaxf(v, __shfl_xor(v, off, 32));
  return v;
}
__device__ inline float waveSum(float v) {
#pragma unroll
  for (int off = 16; off > 0; off >>= 1) v += __shfl_xor(v, off, 32);
  return v;
}

__global__ void score_softmax(const float* __restrict__ enc_p,   // [B,Te,U]
                              const float* __restrict__ dec_p,   // [B,Td,U]
                              const float* __restrict__ Vw,      // [U]
                              const float* __restrict__ Vb,      // [1]
                              float* __restrict__ attn) {        // [B,Td,Te]
  const int bt  = blockIdx.x;          // b*Td + t
  const int b   = bt >> 8;             // Td = 256
  const int tid = threadIdx.x;

  __shared__ __align__(16) float decl[U_];
  __shared__ float vsh[U_];
  __shared__ float red[8];

  // ---- TDM: DMA dec_p[b,t,0:128] (512 B) into LDS. Wave 0 issues once. ----
  if (tid < 32) {
    const unsigned long long ga =
        (unsigned long long)(uintptr_t)(dec_p + (size_t)bt * U_);
    const unsigned int lds = (unsigned int)(uintptr_t)(void*)decl;
    u32x4 g0;
    g0[0] = 1u;                                     // count=1, user mode
    g0[1] = lds;                                    // D#.lds_addr (bytes)
    g0[2] = (unsigned int)(ga & 0xFFFFFFFFu);       // global_addr[31:0]
    g0[3] = (unsigned int)((ga >> 32) & 0x01FFFFFFu) | (2u << 30);  // [56:32] | type=2
    i32x8 g1;
    g1[0] = (int)(2u << 16);        // data_size = 4 B
    g1[1] = (int)(128u << 16);      // tensor_dim0 = 128 (bits 63:48)
    g1[2] = (int)(1024u << 16);     // tensor_dim1 = B*Td = 1024 (bits 95:80)
    g1[3] = (int)(128u << 16);      // tile_dim0 = 128 (bits 127:112)
    g1[4] = 1;                      // tile_dim1 = 1
    g1[5] = 128;                    // tensor_dim0_stride = 128
    g1[6] = 0;
    g1[7] = 0;
    i32x4 z4 = {0, 0, 0, 0};
#if __clang_major__ >= 23
    i32x8 z8 = {0, 0, 0, 0, 0, 0, 0, 0};
    __builtin_amdgcn_tensor_load_to_lds(g0, g1, z4, z4, z8, 0);
#else
    __builtin_amdgcn_tensor_load_to_lds(g0, g1, z4, z4, 0);
#endif
  }
  if (tid < U_) vsh[tid] = Vw[tid];
  __builtin_amdgcn_s_wait_tensorcnt(0);   // no-op for waves that issued nothing
  __syncthreads();

  const float vb = Vb[0];
  const float* erow_base = enc_p + (size_t)b * TE_ * U_;

  float s[2];
#pragma unroll
  for (int rep = 0; rep < 2; ++rep) {
    const int e = tid + rep * 256;
    const float* er = erow_base + (size_t)e * U_;
    __builtin_prefetch(er + U_, 0, 0);           // global_prefetch_b8
    float acc = 0.0f;
#pragma unroll 4
    for (int u = 0; u < U_; u += 4) {
      f32x4 ev = *(const f32x4*)(er + u);
      acc += tanhf(ev[0] + decl[u + 0]) * vsh[u + 0];
      acc += tanhf(ev[1] + decl[u + 1]) * vsh[u + 1];
      acc += tanhf(ev[2] + decl[u + 2]) * vsh[u + 2];
      acc += tanhf(ev[3] + decl[u + 3]) * vsh[u + 3];
    }
    s[rep] = acc + vb;
  }

  // block max (wave32 shuffles + 8 partials in LDS)
  float m = waveMax(fmaxf(s[0], s[1]));
  if ((tid & 31) == 0) red[tid >> 5] = m;
  __syncthreads();
  float bmax = red[0];
#pragma unroll
  for (int i = 1; i < 8; ++i) bmax = fmaxf(bmax, red[i]);
  __syncthreads();

  const float p0 = __expf(s[0] - bmax);
  const float p1 = __expf(s[1] - bmax);
  float lsum = waveSum(p0 + p1);
  if ((tid & 31) == 0) red[tid >> 5] = lsum;
  __syncthreads();
  float bsum = red[0];
#pragma unroll
  for (int i = 1; i < 8; ++i) bsum += red[i];
  const float inv = 1.0f / bsum;

  attn[(size_t)bt * TE_ + tid]       = p0 * inv;
  attn[(size_t)bt * TE_ + tid + 256] = p1 * inv;
}

// ---------------------------------------------------------------------------
extern "C" void kernel_launch(void* const* d_in, const int* in_sizes, int n_in,
                              void* d_out, int out_size, void* d_ws, size_t ws_size,
                              hipStream_t stream) {
  const float* enc = (const float*)d_in[0];  // [B,Te,De]
  const float* dec = (const float*)d_in[1];  // [B,Td,Dd]
  const float* W1w = (const float*)d_in[2];  // [De,U]
  const float* W1b = (const float*)d_in[3];  // [U]
  const float* W2w = (const float*)d_in[4];  // [Dd,U]
  const float* W2b = (const float*)d_in[5];  // [U]
  const float* Vw  = (const float*)d_in[6];  // [U,1]
  const float* Vb  = (const float*)d_in[7];  // [1]

  float* context = (float*)d_out;                                   // [B,Td,De]
  float* attn    = context + (size_t)B_ * TD_ * DE_;                // [B,Td,Te]

  float* encp = (float*)d_ws;                                       // [B,Te,U]
  float* decp = encp + (size_t)B_ * TE_ * U_;                       // [B,Td,U]

  // 1) enc_p = enc @ W1 + b1   (M=2048, K=512, N=128)
  gemm_wmma_f16<DE_><<<dim3((B_ * TE_) / 128, U_ / 16, 1), 256, 0, stream>>>(
      enc, W1w, W1b, encp, U_, 0, 0, 0);

  // 2) dec_p = dec @ W2 + b2   (M=1024, K=512, N=128)
  gemm_wmma_f16<DE_><<<dim3((B_ * TD_) / 128, U_ / 16, 1), 256, 0, stream>>>(
      dec, W2w, W2b, decp, U_, 0, 0, 0);

  // 3+4) score -> softmax -> attn weights (written to final output slot)
  score_softmax<<<dim3(B_ * TD_), 256, 0, stream>>>(encp, decp, Vw, Vb, attn);

  // 5) context[b] = attn[b] @ enc[b]   (M=256, N=512, K=512, batched over B)
  gemm_wmma_f16<TE_><<<dim3(TD_ / 128, DE_ / 16, B_), 256, 0, stream>>>(
      attn, enc, nullptr, context, DE_,
      (size_t)TD_ * TE_, (size_t)TE_ * DE_, (size_t)TD_ * DE_);
}
